// Attention_Critic_87273735455167
// MI455X (gfx1250) — matrix operand order repaired and checked by reading, not compiled, not run on hardware
//
#include <hip/hip_runtime.h>
#include <hip/hip_bf16.h>

// ---------------------------------------------------------------------------
// Fused Attention-Critic for gfx1250 (MI455X), wave32 + WMMA bf16.
// B=32768, A=8, S=256, D=256, H=256, NH=4, HD=64.
// One workgroup = 16 batch items, 256 threads = 8 wave32s.
// GEMMs are N-split across waves (each wave owns 2 column tiles for ALL row
// tiles) so every pre-swizzled weight fragment is read from L2 exactly once
// per workgroup. Both B fragments (global/L2) and A fragments (LDS) are
// register double-buffered one step ahead of the WMMAs that consume them.
// ---------------------------------------------------------------------------

typedef __bf16 bf16_t;
typedef __attribute__((ext_vector_type(16))) __bf16 v16bf;
typedef __attribute__((ext_vector_type(8)))  __bf16 v8bf;
typedef __attribute__((ext_vector_type(8)))  float  v8f;

#define BT 16            // batch items per workgroup
#define NTHREADS 256

// ---- LDS layout (bytes), total 230400 (< 320KB WGP budget) ----------------
// [0,      57344)  : sn (stage1-2)  -> Kb  112x256 bf16 (stage3+)
// [57344,  65536)  : (sn tail)      -> av  16x256 bf16  (stage4+)
// [65536, 131072)  : enc 128x256 bf16 (stage2-3) -> qbuf 16x256 f32 (stage6+)
// [131072,147456)  : Qf 16x256 f32
// [147456,204800)  : Vb 112x256 bf16
// [204800,221184)  : x  16x512 bf16
// [221184,228352)  : attention partial dots 64*28 f32
// [228352,230400)  : attention weights      64*8  f32
#define SMEM_BYTES 230400

// ---- WMMA fragment helpers (layouts per CDNA5 ISA 7.12.2) -----------------

// 16-bit A matrix 16x32: lane l holds row M=l&15; elements e=0..7 -> K=(l>>4)*8+e,
// e=8..15 -> K=16+(l>>4)*8+(e-8). Two contiguous 16B chunks per lane.
__device__ __forceinline__ v16bf load_a_frag(const bf16_t* tile_base, int stride,
                                             int kb, int lane) {
  int m  = lane & 15;
  int hi = lane >> 4;
  const bf16_t* p = tile_base + m * stride + kb * 32 + hi * 8;
  v8bf lo = *(const v8bf*)(p);
  v8bf hh = *(const v8bf*)(p + 16);
  v16bf f;
#pragma unroll
  for (int i = 0; i < 8; ++i) { f[i] = lo[i]; f[8 + i] = hh[i]; }
  return f;
}

// B fragments are pre-swizzled in global memory: frag (kt,nt), lane l owns 16
// contiguous bf16 (32B). Packed index: ((kt*NT+nt)*32 + lane)*16.
__device__ __forceinline__ v16bf load_b_frag(const bf16_t* pack, int kt, int nt,
                                             int NT, int lane) {
  return *(const v16bf*)(pack + (((size_t)(kt * NT + nt) * 32) + lane) * 16);
}

__device__ __forceinline__ v8f wmma_bf16(v16bf a, v16bf b, v8f c) {
  return __builtin_amdgcn_wmma_f32_16x16x32_bf16(false, a, false, b, (short)0, c,
                                                 false, false);
}

__device__ __forceinline__ float wave_reduce_add(float v) {
#pragma unroll
  for (int m = 16; m >= 1; m >>= 1) v += __shfl_xor(v, m, 32);
  return v;
}

// ---------------------------------------------------------------------------
// Prep: pack a row-major (K x 256) f32 weight matrix into bf16 WMMA-B
// fragment order. lane l element e of frag (kt,nt): K=kt*32+(e>>3)*16+(l>>4)*8+(e&7),
// N=nt*16+(l&15).
// ---------------------------------------------------------------------------
__global__ __launch_bounds__(256) void pack_weight_kernel(const float* __restrict__ W,
                                                          bf16_t* __restrict__ dst,
                                                          int K) {
  int t = blockIdx.x * 256 + threadIdx.x;
  if (t >= K * 256) return;
  int e    = t & 15;
  int lane = (t >> 4) & 31;
  int fidx = t >> 9;
  int nt = fidx & 15;
  int kt = fidx >> 4;
  int k = kt * 32 + ((e >> 3) * 16) + ((lane >> 4) * 8) + (e & 7);
  int n = nt * 16 + (lane & 15);
  dst[t] = (bf16_t)W[(size_t)k * 256 + n];
}

// ---------------------------------------------------------------------------
// Main fused kernel.
// ---------------------------------------------------------------------------
__global__ __launch_bounds__(NTHREADS) void attn_critic_kernel(
    const float* __restrict__ s,   // (B, 2048)
    const float* __restrict__ be,  // (256)
    const float* __restrict__ bv,  // (256)
    const float* __restrict__ bo,  // (256)
    const float* __restrict__ b1,  // (256)
    const float* __restrict__ W3,  // (256)
    const float* __restrict__ b3,  // (1)
    const bf16_t* __restrict__ WeP, const bf16_t* __restrict__ WqP,
    const bf16_t* __restrict__ WkP, const bf16_t* __restrict__ WvP,
    const bf16_t* __restrict__ WoP, const bf16_t* __restrict__ W1P,
    float* __restrict__ out)       // (B)
{
  __shared__ __attribute__((aligned(32))) char smem[SMEM_BYTES];
  bf16_t* sn   = (bf16_t*)(smem + 0);        // 128x256 bf16 (stages 1-2)
  bf16_t* Kb   = (bf16_t*)(smem + 0);        // 112x256 bf16 (stage 3+)
  bf16_t* avb  = (bf16_t*)(smem + 57344);    // 16x256 bf16  (stage 4+)
  bf16_t* encb = (bf16_t*)(smem + 65536);    // 128x256 bf16 (stages 2-3)
  float*  qbuf = (float*) (smem + 65536);    // 16x256 f32   (stage 6+)
  float*  Qf   = (float*) (smem + 131072);   // 16x256 f32
  bf16_t* Vb   = (bf16_t*)(smem + 147456);   // 112x256 bf16
  bf16_t* xb   = (bf16_t*)(smem + 204800);   // 16x512 bf16
  float*  part = (float*) (smem + 221184);   // 64*28 f32
  float*  attw = (float*) (smem + 228352);   // 64*8  f32

  const int tid  = threadIdx.x;
  const int lane = tid & 31;
  const int wave = tid >> 5;
  const int ib0  = blockIdx.x * BT;
  const int nt0  = wave * 2;                 // this wave's two N-tiles
  const int nt1  = nt0 + 1;
  const int hi   = lane >> 4;
  const int nlo  = lane & 15;

  // ---------------- Stage 1: LayerNorm over 2048, write sn (bf16) ----------
  // Wave w handles items 2w, 2w+1. sn row layout: (item*8 + agent) x 256.
  // Input is a single-use 268MB stream: keep it out of L2 (non-temporal).
  for (int rr = 0; rr < 2; ++rr) {
    const int li = wave * 2 + rr;
    const float* rp = s + (size_t)(ib0 + li) * 2048;
    float vals[64];
    float sum = 0.f, ss = 0.f;
#pragma unroll
    for (int j = 0; j < 64; ++j) {
      float v = __builtin_nontemporal_load(rp + lane + 32 * j);
      vals[j] = v; sum += v; ss += v * v;
    }
    sum = wave_reduce_add(sum);
    ss  = wave_reduce_add(ss);
    const float mean = sum * (1.0f / 2048.0f);
    const float var  = ss * (1.0f / 2048.0f) - mean * mean;
    const float rs   = rsqrtf(var + 1e-5f);
#pragma unroll
    for (int j = 0; j < 64; ++j) {
      int idx = lane + 32 * j;
      int a = idx >> 8, off = idx & 255;
      sn[(li * 8 + a) * 256 + off] = (bf16_t)((vals[j] - mean) * rs);
    }
  }
  __syncthreads();

  // ---------------- Stage 2: enc = sn @ We + be  (128x256x256) -------------
  // N-split, B double-buffered over kb (global/L2), A double-buffered over mt
  // (LDS) so every s_wait covers a load issued one 2-WMMA group earlier.
  {
    v8f acc[8][2] = {};
    v16bf b0 = load_b_frag(WeP, 0, nt0, 16, lane);
    v16bf b1 = load_b_frag(WeP, 0, nt1, 16, lane);
    v16bf a  = load_a_frag(sn, 256, 0, lane);
    for (int kb = 0; kb < 8; ++kb) {
      v16bf nb0 = load_b_frag(WeP, (kb + 1) & 7, nt0, 16, lane);
      v16bf nb1 = load_b_frag(WeP, (kb + 1) & 7, nt1, 16, lane);
#pragma unroll
      for (int mt = 0; mt < 8; ++mt) {
        v16bf na = (mt < 7) ? load_a_frag(sn + (mt + 1) * 16 * 256, 256, kb, lane)
                            : load_a_frag(sn, 256, (kb + 1) & 7, lane);
        acc[mt][0] = wmma_bf16(a, b0, acc[mt][0]);
        acc[mt][1] = wmma_bf16(a, b1, acc[mt][1]);
        a = na;
      }
      b0 = nb0; b1 = nb1;
    }
    // Epilogue: +be, bf16 store with query/key row re-layout.
#pragma unroll
    for (int nn = 0; nn < 2; ++nn) {
      const int col = (nt0 + nn) * 16 + nlo;
      const float bias = be[col];
#pragma unroll
      for (int mt = 0; mt < 8; ++mt) {
#pragma unroll
        for (int r = 0; r < 8; ++r) {
          const int row = mt * 16 + hi * 8 + r;
          const int li = row >> 3, ag = row & 7;
          const float v = acc[mt][nn][r] + bias;
          const int erow = (ag == 0) ? li : (16 + li * 7 + (ag - 1));
          encb[erow * 256 + col] = (bf16_t)v;
          if (ag == 0) xb[li * 512 + col] = (bf16_t)v;   // s_i half of x
        }
      }
    }
  }
  __syncthreads();

  // ---------------- Stage 3: Q/K/V GEMMs (N-split) -------------------------
  // Q = encb[0:16] @ Wq
  {
    v8f acc0 = {}, acc1 = {};
    v16bf b0 = load_b_frag(WqP, 0, nt0, 16, lane);
    v16bf b1 = load_b_frag(WqP, 0, nt1, 16, lane);
    v16bf a  = load_a_frag(encb, 256, 0, lane);
    for (int kb = 0; kb < 8; ++kb) {
      v16bf nb0 = load_b_frag(WqP, (kb + 1) & 7, nt0, 16, lane);
      v16bf nb1 = load_b_frag(WqP, (kb + 1) & 7, nt1, 16, lane);
      v16bf na  = load_a_frag(encb, 256, (kb + 1) & 7, lane);
      acc0 = wmma_bf16(a, b0, acc0);
      acc1 = wmma_bf16(a, b1, acc1);
      a = na; b0 = nb0; b1 = nb1;
    }
#pragma unroll
    for (int nn = 0; nn < 2; ++nn) {
      const int col = (nt0 + nn) * 16 + nlo;
      const v8f& acc = nn ? acc1 : acc0;
#pragma unroll
      for (int r = 0; r < 8; ++r) Qf[(hi * 8 + r) * 256 + col] = acc[r];
    }
  }
  // K = encb[16:128] @ Wk  (7 M-tiles)
  {
    v8f acc[7][2] = {};
    v16bf b0 = load_b_frag(WkP, 0, nt0, 16, lane);
    v16bf b1 = load_b_frag(WkP, 0, nt1, 16, lane);
    v16bf a  = load_a_frag(encb + 16 * 256, 256, 0, lane);
    for (int kb = 0; kb < 8; ++kb) {
      v16bf nb0 = load_b_frag(WkP, (kb + 1) & 7, nt0, 16, lane);
      v16bf nb1 = load_b_frag(WkP, (kb + 1) & 7, nt1, 16, lane);
#pragma unroll
      for (int mt = 0; mt < 7; ++mt) {
        v16bf na = (mt < 6)
            ? load_a_frag(encb + (16 + (mt + 1) * 16) * 256, 256, kb, lane)
            : load_a_frag(encb + 16 * 256, 256, (kb + 1) & 7, lane);
        acc[mt][0] = wmma_bf16(a, b0, acc[mt][0]);
        acc[mt][1] = wmma_bf16(a, b1, acc[mt][1]);
        a = na;
      }
      b0 = nb0; b1 = nb1;
    }
#pragma unroll
    for (int nn = 0; nn < 2; ++nn) {
      const int col = (nt0 + nn) * 16 + nlo;
#pragma unroll
      for (int mt = 0; mt < 7; ++mt)
#pragma unroll
        for (int r = 0; r < 8; ++r)
          Kb[(mt * 16 + hi * 8 + r) * 256 + col] = (bf16_t)acc[mt][nn][r];
    }
  }
  // V = leaky_relu(encb[16:128] @ Wv + bv)
  {
    v8f acc[7][2] = {};
    v16bf b0 = load_b_frag(WvP, 0, nt0, 16, lane);
    v16bf b1 = load_b_frag(WvP, 0, nt1, 16, lane);
    v16bf a  = load_a_frag(encb + 16 * 256, 256, 0, lane);
    for (int kb = 0; kb < 8; ++kb) {
      v16bf nb0 = load_b_frag(WvP, (kb + 1) & 7, nt0, 16, lane);
      v16bf nb1 = load_b_frag(WvP, (kb + 1) & 7, nt1, 16, lane);
#pragma unroll
      for (int mt = 0; mt < 7; ++mt) {
        v16bf na = (mt < 6)
            ? load_a_frag(encb + (16 + (mt + 1) * 16) * 256, 256, kb, lane)
            : load_a_frag(encb + 16 * 256, 256, (kb + 1) & 7, lane);
        acc[mt][0] = wmma_bf16(a, b0, acc[mt][0]);
        acc[mt][1] = wmma_bf16(a, b1, acc[mt][1]);
        a = na;
      }
      b0 = nb0; b1 = nb1;
    }
#pragma unroll
    for (int nn = 0; nn < 2; ++nn) {
      const int col = (nt0 + nn) * 16 + nlo;
      const float bias = bv[col];
#pragma unroll
      for (int mt = 0; mt < 7; ++mt) {
#pragma unroll
        for (int r = 0; r < 8; ++r) {
          float v = acc[mt][nn][r] + bias;
          v = v > 0.f ? v : 0.01f * v;                    // leaky_relu
          Vb[(mt * 16 + hi * 8 + r) * 256 + col] = (bf16_t)v;
        }
      }
    }
  }
  __syncthreads();

  // ---------------- Stage 4: attention (16 items x 4 heads, 7 keys) --------
  {
    const int p = tid >> 2, sub = tid & 3;   // 64 (item,head) pairs x 4 sublanes
    const int li = p >> 2, head = p & 3;
    const float*  qrow = Qf + li * 256 + head * 64;
    const bf16_t* krow = Kb + (li * 7) * 256 + head * 64;
#pragma unroll
    for (int k = 0; k < 7; ++k) {
      float d = 0.f;
#pragma unroll
      for (int dd = 0; dd < 16; ++dd)
        d += qrow[sub * 16 + dd] * (float)krow[k * 256 + sub * 16 + dd];
      part[p * 28 + k * 4 + sub] = d;
    }
    __syncthreads();
    if (sub == 0) {
      float sc[7], mx = -1e30f;
#pragma unroll
      for (int k = 0; k < 7; ++k) {
        float d = part[p * 28 + k * 4] + part[p * 28 + k * 4 + 1] +
                  part[p * 28 + k * 4 + 2] + part[p * 28 + k * 4 + 3];
        sc[k] = d * 0.125f;                  // 1/sqrt(64)
        mx = fmaxf(mx, sc[k]);
      }
      float tot = 0.f;
#pragma unroll
      for (int k = 0; k < 7; ++k) { sc[k] = __expf(sc[k] - mx); tot += sc[k]; }
      const float inv = 1.0f / tot;
#pragma unroll
      for (int k = 0; k < 7; ++k) attw[p * 8 + k] = sc[k] * inv;
    }
    __syncthreads();
    const bf16_t* vrow = Vb + (li * 7) * 256 + head * 64;
    float w[7];
#pragma unroll
    for (int k = 0; k < 7; ++k) w[k] = attw[p * 8 + k];
#pragma unroll
    for (int dd = 0; dd < 16; ++dd) {
      float a = 0.f;
#pragma unroll
      for (int k = 0; k < 7; ++k)
        a += w[k] * (float)vrow[k * 256 + sub * 16 + dd];
      avb[li * 256 + head * 64 + sub * 16 + dd] = (bf16_t)a;
    }
  }
  __syncthreads();

  // ---------------- Stage 5: out = av @ Wo + bo -> x[:,256:512] ------------
  {
    v8f acc0 = {}, acc1 = {};
    v16bf b0 = load_b_frag(WoP, 0, nt0, 16, lane);
    v16bf b1 = load_b_frag(WoP, 0, nt1, 16, lane);
    v16bf a  = load_a_frag(avb, 256, 0, lane);
    for (int kb = 0; kb < 8; ++kb) {
      v16bf nb0 = load_b_frag(WoP, (kb + 1) & 7, nt0, 16, lane);
      v16bf nb1 = load_b_frag(WoP, (kb + 1) & 7, nt1, 16, lane);
      v16bf na  = load_a_frag(avb, 256, (kb + 1) & 7, lane);
      acc0 = wmma_bf16(a, b0, acc0);
      acc1 = wmma_bf16(a, b1, acc1);
      a = na; b0 = nb0; b1 = nb1;
    }
#pragma unroll
    for (int nn = 0; nn < 2; ++nn) {
      const int col = (nt0 + nn) * 16 + nlo;
      const float bias = bo[col];
      const v8f& acc = nn ? acc1 : acc0;
#pragma unroll
      for (int r = 0; r < 8; ++r)
        xb[(hi * 8 + r) * 512 + 256 + col] = (bf16_t)(acc[r] + bias);
    }
  }
  __syncthreads();

  // ---------------- Stage 6: q = relu(x @ W1 + b1)  (K=512) ----------------
  {
    v8f acc0 = {}, acc1 = {};
    v16bf b0 = load_b_frag(W1P, 0, nt0, 16, lane);
    v16bf b1 = load_b_frag(W1P, 0, nt1, 16, lane);
    v16bf a  = load_a_frag(xb, 512, 0, lane);
    for (int kb = 0; kb < 16; ++kb) {
      v16bf nb0 = load_b_frag(W1P, (kb + 1) & 15, nt0, 16, lane);
      v16bf nb1 = load_b_frag(W1P, (kb + 1) & 15, nt1, 16, lane);
      v16bf na  = load_a_frag(xb, 512, (kb + 1) & 15, lane);
      acc0 = wmma_bf16(a, b0, acc0);
      acc1 = wmma_bf16(a, b1, acc1);
      a = na; b0 = nb0; b1 = nb1;
    }
#pragma unroll
    for (int nn = 0; nn < 2; ++nn) {
      const int col = (nt0 + nn) * 16 + nlo;
      const float bias = b1[col];
      const v8f& acc = nn ? acc1 : acc0;
#pragma unroll
      for (int r = 0; r < 8; ++r) {
        float v = acc[r] + bias;
        qbuf[(hi * 8 + r) * 256 + col] = v > 0.f ? v : 0.f;
      }
    }
  }
  __syncthreads();

  // ---------------- Stage 7: LN(q) @ W3 + b3 -------------------------------
  for (int rr = 0; rr < 2; ++rr) {
    const int li = wave * 2 + rr;
    const float* qr = qbuf + li * 256;
    float v[8];
    float sum = 0.f, ss = 0.f;
#pragma unroll
    for (int j = 0; j < 8; ++j) {
      float x = qr[lane + 32 * j];
      v[j] = x; sum += x; ss += x * x;
    }
    sum = wave_reduce_add(sum);
    ss  = wave_reduce_add(ss);
    const float mean = sum * (1.0f / 256.0f);
    const float var  = ss * (1.0f / 256.0f) - mean * mean;
    const float rs   = rsqrtf(var + 1e-5f);
    float acc = 0.f;
#pragma unroll
    for (int j = 0; j < 8; ++j) {
      const int col = lane + 32 * j;
      acc += (v[j] - mean) * rs * W3[col];
    }
    acc = wave_reduce_add(acc);
    if (lane == 0) out[ib0 + li] = acc + b3[0];
  }
}

// ---------------------------------------------------------------------------
extern "C" void kernel_launch(void* const* d_in, const int* in_sizes, int n_in,
                              void* d_out, int out_size, void* d_ws, size_t ws_size,
                              hipStream_t stream) {
  (void)in_sizes; (void)n_in; (void)out_size; (void)ws_size;
  const float* s  = (const float*)d_in[0];
  const float* We = (const float*)d_in[1];
  const float* be = (const float*)d_in[2];
  const float* Wq = (const float*)d_in[3];
  const float* Wk = (const float*)d_in[4];
  const float* Wv = (const float*)d_in[5];
  const float* bv = (const float*)d_in[6];
  const float* Wo = (const float*)d_in[7];
  const float* bo = (const float*)d_in[8];
  const float* W1 = (const float*)d_in[9];
  const float* b1 = (const float*)d_in[10];
  const float* W3 = (const float*)d_in[11];
  const float* b3 = (const float*)d_in[12];
  float* out = (float*)d_out;

  // Packed bf16 weights in workspace (elements, each 256 cols):
  bf16_t* ws  = (bf16_t*)d_ws;
  bf16_t* WeP = ws + 0;
  bf16_t* WqP = ws + 65536;
  bf16_t* WkP = ws + 131072;
  bf16_t* WvP = ws + 196608;
  bf16_t* WoP = ws + 262144;
  bf16_t* W1P = ws + 327680;  // 512x256

  pack_weight_kernel<<<256, 256, 0, stream>>>(We, WeP, 256);
  pack_weight_kernel<<<256, 256, 0, stream>>>(Wq, WqP, 256);
  pack_weight_kernel<<<256, 256, 0, stream>>>(Wk, WkP, 256);
  pack_weight_kernel<<<256, 256, 0, stream>>>(Wv, WvP, 256);
  pack_weight_kernel<<<256, 256, 0, stream>>>(Wo, WoP, 256);
  pack_weight_kernel<<<512, 256, 0, stream>>>(W1, W1P, 512);

  attn_critic_kernel<<<32768 / BT, NTHREADS, 0, stream>>>(
      s, be, bv, bo, b1, W3, b3, WeP, WqP, WkP, WvP, WoP, W1P, out);
}
